// VAE_86835648790568
// MI455X (gfx1250) — compile-verified, matrix operand order checked by reading, not run on hardware
//
#include <hip/hip_runtime.h>
#include <hip/hip_bf16.h>
#include <cstdint>

// ---------------------------------------------------------------------------
// Conditional VAE forward on gfx1250 (MI455X), wave32 + WMMA f16->f32.
// fc1 uses async global->LDS double-buffered A-tile staging (ASYNCcnt path);
// fc2/fc3 are register-direct WMMA GEMMs; convs are fused per-sample LDS conv.
// ---------------------------------------------------------------------------

typedef __attribute__((ext_vector_type(16))) _Float16 v16h;
typedef __attribute__((ext_vector_type(8)))  _Float16 v8h;
typedef __attribute__((ext_vector_type(8)))  float    v8f;
typedef int v4i_ls __attribute__((vector_size(16)));   // async-copy builtin type

#define BATCH 8192

// Padded K dims (multiples of 32 so every WMMA K-step is full)
#define KP1 800   // fc1 K: 784 -> 800
#define KP2 416   // fc2 K: 400 -> 416
#define KP3 32    // fc3 K: 30  -> 32

// Workspace byte offsets (all 32B-aligned)
#define OFF_H      0                                   // fp16 [B][800]
#define OFF_W1     (OFF_H   + (size_t)BATCH*KP1*2)     // fp16 [400][800]
#define OFF_H1     (OFF_W1  + (size_t)400*KP1*2)       // fp16 [B][416]
#define OFF_W2     (OFF_H1  + (size_t)BATCH*KP2*2)     // fp16 [48][416]
#define OFF_Z      (OFF_W2  + (size_t)48*KP2*2)        // fp16 [B][32]
#define OFF_W3     (OFF_Z   + (size_t)BATCH*KP3*2)     // fp16 [400][32]
#define OFF_H3     (OFF_W3  + (size_t)400*KP3*2)       // f32  [B][392]

// ---------------------------------------------------------------------------
// CDNA5 async global->LDS copy (ASYNCcnt) with __has_builtin hedging.
// Builtin signature (from hipcc diagnostic): (int4 AS1*, int4 AS3*, Imm, Imm).
// ---------------------------------------------------------------------------
__device__ __forceinline__ void async_copy_b128(const _Float16* g, _Float16* l) {
#if __has_builtin(__builtin_amdgcn_global_load_async_to_lds_b128)
  __builtin_amdgcn_global_load_async_to_lds_b128(
      (__attribute__((address_space(1))) v4i_ls*)g,
      (__attribute__((address_space(3))) v4i_ls*)l, 0, 0);
#else
  unsigned lds_off = (unsigned)(uintptr_t)(__attribute__((address_space(3))) _Float16*)l;
  asm volatile("global_load_async_to_lds_b128 %0, %1, off"
               :: "v"(lds_off), "v"(g) : "memory");
#endif
}

template <int N>
__device__ __forceinline__ void async_wait() {
#if __has_builtin(__builtin_amdgcn_s_wait_asynccnt)
  __builtin_amdgcn_s_wait_asynccnt(N);
#else
  asm volatile("s_wait_asynccnt %0" :: "n"(N) : "memory");
#endif
}

// ---------------------------------------------------------------------------
// WMMA fragment loaders.  16-bit A (16x32) / B (32x16 as [N][K] rows) per-lane
// layout (ISA 7.12.2): lanes 0-15 hold row (lane), K = {kb..kb+7, kb+16..kb+23};
// lanes 16-31 hold row (lane-16), K = {kb+8..kb+15, kb+24..kb+31}.
// Two contiguous 16B chunks per lane -> b128 loads (global or ds).
// ---------------------------------------------------------------------------
__device__ __forceinline__ v16h load_frag16(const _Float16* __restrict__ base,
                                            int row0, int ldk, int kb, int lane) {
  int r  = row0 + (lane & 15);
  int ko = kb + ((lane >> 4) << 3);            // +0 or +8
  const _Float16* p = base + (size_t)r * ldk + ko;
  v8h lo = *(const v8h*)(p);
  v8h hi = *(const v8h*)(p + 16);
  v16h f;
#pragma unroll
  for (int i = 0; i < 8; ++i) { f[i] = lo[i]; f[i + 8] = hi[i]; }
  return f;
}

__device__ __forceinline__ v16h load_frag16_lds(const _Float16* base,
                                                int row0, int lane) {
  int r  = row0 + (lane & 15);
  int ko = (lane >> 4) << 3;
  const _Float16* p = base + r * 32 + ko;
  v8h lo = *(const v8h*)(p);
  v8h hi = *(const v8h*)(p + 16);
  v16h f;
#pragma unroll
  for (int i = 0; i < 8; ++i) { f[i] = lo[i]; f[i + 8] = hi[i]; }
  return f;
}

// ---------------------------------------------------------------------------
// K0: convert/pad weights to fp16, zero h1 K-pad columns.
// ---------------------------------------------------------------------------
__global__ void vae_prep_kernel(const float* __restrict__ fc1_w,
                                const float* __restrict__ fc21_w,
                                const float* __restrict__ fc22_w,
                                const float* __restrict__ fc3_w,
                                _Float16* __restrict__ w1h,
                                _Float16* __restrict__ w2h,
                                _Float16* __restrict__ w3h,
                                _Float16* __restrict__ h1) {
  const int N1 = 400 * KP1;        // 320000
  const int N2 = 48  * KP2;        // 19968
  const int N3 = 400 * KP3;        // 12800
  const int N4 = BATCH * 16;       // h1 pad cols 400..415
  int total = N1 + N2 + N3 + N4;
  for (int idx = blockIdx.x * blockDim.x + threadIdx.x; idx < total;
       idx += gridDim.x * blockDim.x) {
    if (idx < N1) {
      int n = idx / KP1, k = idx % KP1;
      float v = (k < 784) ? fc1_w[n * 784 + k] : 0.f;
      w1h[idx] = (_Float16)v;
    } else if (idx < N1 + N2) {
      int t = idx - N1;
      int n = t / KP2, k = t % KP2;
      float v = 0.f;
      if (k < 400) {
        if (n < 20)       v = fc21_w[n * 400 + k];
        else if (n < 40)  v = fc22_w[(n - 20) * 400 + k];
      }
      w2h[t] = (_Float16)v;
    } else if (idx < N1 + N2 + N3) {
      int t = idx - N1 - N2;
      int n = t / KP3, k = t % KP3;
      float v = (n < 392 && k < 30) ? fc3_w[n * 30 + k] : 0.f;
      w3h[t] = (_Float16)v;
    } else {
      int t = idx - N1 - N2 - N3;
      int r = t / 16, k = 400 + (t % 16);
      h1[(size_t)r * KP2 + k] = (_Float16)0.f;
    }
  }
}

// ---------------------------------------------------------------------------
// K1: per-sample encoder.  Builds the interleaved [pix|onehot] 11-ch input
// (the reference's resize_ trick), runs conv0(11->3)+relu, conv1(3->1)+relu,
// stores fp16 activation row [800] (zero-padded).
// ---------------------------------------------------------------------------
__global__ void vae_encode_kernel(const float* __restrict__ x,
                                  const long long* __restrict__ cls,
                                  const float* __restrict__ w0g,   // [3][11][3][3]
                                  const float* __restrict__ w1g,   // [1][3][3][3]
                                  _Float16* __restrict__ hout) {   // [B][800]
  __shared__ float ein[11 * 784];
  __shared__ float c0s[3 * 784];
  __shared__ float w0[297];
  __shared__ float w1[27];

  const int b   = blockIdx.x;
  const int tid = threadIdx.x;
  const long long cb = cls[b];
  const float* xb = x + (size_t)b * 784;

  for (int i = tid; i < 297; i += blockDim.x) w0[i] = w0g[i];
  if (tid < 27) w1[tid] = w1g[tid];

  // scrambled layout: e_input[ci][s] = flat[k = ci*784+s], pos=k/11, j=k%11
  for (int idx = tid; idx < 11 * 784; idx += blockDim.x) {
    int pos = idx / 11, j = idx % 11;
    float v = (j == 0) ? xb[pos] : ((cb == (long long)(j - 1)) ? 1.f : 0.f);
    ein[idx] = v;
  }
  __syncthreads();

  // conv0: 11 -> 3, 3x3, pad 1, relu
  for (int oi = tid; oi < 3 * 784; oi += blockDim.x) {
    int o = oi / 784, s = oi % 784, hh = s / 28, ww = s % 28;
    float acc = 0.f;
#pragma unroll
    for (int ic = 0; ic < 11; ++ic) {
      const float* wp = &w0[(o * 11 + ic) * 9];
#pragma unroll
      for (int kh = 0; kh < 3; ++kh) {
        int ih = hh + kh - 1;
        if (ih < 0 || ih >= 28) continue;
#pragma unroll
        for (int kw = 0; kw < 3; ++kw) {
          int iw = ww + kw - 1;
          if (iw < 0 || iw >= 28) continue;
          acc += ein[ic * 784 + ih * 28 + iw] * wp[kh * 3 + kw];
        }
      }
    }
    c0s[oi] = acc > 0.f ? acc : 0.f;
  }
  __syncthreads();

  // conv1: 3 -> 1, relu, fp16 store
  _Float16* hb = hout + (size_t)b * KP1;
  for (int s = tid; s < 784; s += blockDim.x) {
    int hh = s / 28, ww = s % 28;
    float acc = 0.f;
#pragma unroll
    for (int ic = 0; ic < 3; ++ic) {
      const float* wp = &w1[ic * 9];
#pragma unroll
      for (int kh = 0; kh < 3; ++kh) {
        int ih = hh + kh - 1;
        if (ih < 0 || ih >= 28) continue;
#pragma unroll
        for (int kw = 0; kw < 3; ++kw) {
          int iw = ww + kw - 1;
          if (iw < 0 || iw >= 28) continue;
          acc += c0s[ic * 784 + ih * 28 + iw] * wp[kh * 3 + kw];
        }
      }
    }
    hb[s] = (_Float16)(acc > 0.f ? acc : 0.f);
  }
  if (tid < 16) hb[784 + tid] = (_Float16)0.f;   // K-pad
}

// ---------------------------------------------------------------------------
// K2: fc1 GEMM  [8192 x 784] * [784 x 400]^T + bias, relu -> fp16 [B][416]
// Block = 800 threads = 25 waves, one wave per 16-col N-tile (N=400=25x16).
// Block owns a 64-row M stripe; the 64x32 fp16 A tile per K-step is staged
// into LDS once via GLOBAL_LOAD_ASYNC_TO_LDS_B128 (double-buffered,
// s_wait_asynccnt) and reused by all 25 waves.  128 blocks total.
// ---------------------------------------------------------------------------
__global__ void __launch_bounds__(800)
vae_fc1_kernel(const _Float16* __restrict__ A,   // [B][800]
               const _Float16* __restrict__ Bw,  // [400][800]
               const float* __restrict__ bias,   // [400]
               _Float16* __restrict__ out) {     // [B][416]
  __shared__ _Float16 As[2][64 * 32];            // 2 x 4KB double buffer

  const int tid  = threadIdx.x;
  const int lane = tid & 31;
  const int w    = tid >> 5;                     // 0..24 == N-tile
  const int mbase = blockIdx.x * 64;
  const int nbase = w * 16;

  // stage A tile for K-step kk into buffer buf (256 lanes x b128 = 4KB)
  const bool loader = (tid < 256);
  const int  lrow   = tid >> 2;                  // 0..63
  const int  lseg   = (tid & 3) * 8;             // halves
  const _Float16* gbase = A + (size_t)(mbase + lrow) * KP1 + lseg;

  if (loader) async_copy_b128(gbase + 0 * 32, &As[0][lrow * 32 + lseg]);

  v8f acc[4] = {v8f{}, v8f{}, v8f{}, v8f{}};
  for (int kk = 0; kk < 25; ++kk) {
    int buf = kk & 1;
    if (kk + 1 < 25) {
      if (loader) async_copy_b128(gbase + (kk + 1) * 32,
                                  &As[buf ^ 1][lrow * 32 + lseg]);
      async_wait<1>();      // tile kk resident; tile kk+1 may stay in flight
    } else {
      async_wait<0>();
    }
    __syncthreads();

    int kb = kk * 32;
    __builtin_prefetch(Bw + (size_t)(nbase + (lane & 15)) * KP1 + kb + 32, 0, 3);
    v16h bf = load_frag16(Bw, nbase, KP1, kb, lane);
#pragma unroll
    for (int f = 0; f < 4; ++f) {
      v16h af = load_frag16_lds(&As[buf][0], f * 16, lane);
      acc[f] = __builtin_amdgcn_wmma_f32_16x16x32_f16(
          false, af, false, bf, (short)0, acc[f], false, false);
    }
    __syncthreads();        // reads of As[buf] done before it is overwritten
  }

  int col  = nbase + (lane & 15);
  int rsel = (lane >> 4) << 3;
  float bv = bias[col];
#pragma unroll
  for (int f = 0; f < 4; ++f) {
#pragma unroll
    for (int i = 0; i < 8; ++i) {
      int r = mbase + f * 16 + i + rsel;
      float v = acc[f][i] + bv;
      out[(size_t)r * KP2 + col] = (_Float16)(v > 0.f ? v : 0.f);
    }
  }
}

// ---------------------------------------------------------------------------
// K3: fc21/fc22 fused GEMM  [8192 x 400] * [400 x 40]^T  (N padded to 48)
// mu -> d_out + fp16 z[:,0:20];  logvar -> d_out.  384 waves (48 blocks).
// ---------------------------------------------------------------------------
__global__ void vae_fc2_kernel(const _Float16* __restrict__ A,   // [B][416]
                               const _Float16* __restrict__ Bw,  // [48][416]
                               const float* __restrict__ b21,
                               const float* __restrict__ b22,
                               float* __restrict__ mu_out,       // [B][20]
                               float* __restrict__ lv_out,       // [B][20]
                               _Float16* __restrict__ zbuf) {    // [B][32]
  const int NT = 3;
  int wave = (blockIdx.x * blockDim.x + threadIdx.x) >> 5;
  int lane = threadIdx.x & 31;
  int nt = wave % NT, mt = wave / NT;
  int mbase = mt * 64, nbase = nt * 16;

  v8f acc[4] = {v8f{}, v8f{}, v8f{}, v8f{}};
  for (int kk = 0; kk < 13; ++kk) {
    int kb = kk * 32;
    __builtin_prefetch(A + (size_t)(mbase + (lane & 15)) * KP2 + kb + 32, 0, 3);
    v16h bf = load_frag16(Bw, nbase, KP2, kb, lane);
#pragma unroll
    for (int f = 0; f < 4; ++f) {
      v16h af = load_frag16(A, mbase + f * 16, KP2, kb, lane);
      acc[f] = __builtin_amdgcn_wmma_f32_16x16x32_f16(
          false, af, false, bf, (short)0, acc[f], false, false);
    }
  }

  int col  = nbase + (lane & 15);
  int rsel = (lane >> 4) << 3;
#pragma unroll
  for (int f = 0; f < 4; ++f) {
#pragma unroll
    for (int i = 0; i < 8; ++i) {
      int r = mbase + f * 16 + i + rsel;
      float v = acc[f][i];
      if (col < 20) {
        v += b21[col];
        mu_out[(size_t)r * 20 + col] = v;
        zbuf[(size_t)r * 32 + col]   = (_Float16)v;
      } else if (col < 40) {
        v += b22[col - 20];
        lv_out[(size_t)r * 20 + (col - 20)] = v;
      }
    }
  }
}

// ---------------------------------------------------------------------------
// K4: z one-hot columns 20..29 + zero pad 30..31
// ---------------------------------------------------------------------------
__global__ void vae_zfill_kernel(const long long* __restrict__ cls,
                                 _Float16* __restrict__ zbuf) {
  int total = BATCH * 12;
  for (int idx = blockIdx.x * blockDim.x + threadIdx.x; idx < total;
       idx += gridDim.x * blockDim.x) {
    int b = idx / 12, j = idx % 12;          // z col = 20 + j
    float v = (j < 10 && cls[b] == (long long)j) ? 1.f : 0.f;
    zbuf[(size_t)b * 32 + 20 + j] = (_Float16)v;
  }
}

// ---------------------------------------------------------------------------
// K5: fc3 GEMM  [8192 x 30] * [30 x 392]^T + bias, relu -> f32 [B][392]
// single K-step; N padded to 400 (cols>=392 masked).  3200 waves.
// ---------------------------------------------------------------------------
__global__ void vae_fc3_kernel(const _Float16* __restrict__ A,   // [B][32]
                               const _Float16* __restrict__ Bw,  // [400][32]
                               const float* __restrict__ bias,   // [392]
                               float* __restrict__ out) {        // [B][392]
  const int NT = 25;
  int wave = (blockIdx.x * blockDim.x + threadIdx.x) >> 5;
  int lane = threadIdx.x & 31;
  int nt = wave % NT, mt = wave / NT;
  int mbase = mt * 64, nbase = nt * 16;

  v8f acc[4] = {v8f{}, v8f{}, v8f{}, v8f{}};
  v16h bf = load_frag16(Bw, nbase, KP3, 0, lane);
#pragma unroll
  for (int f = 0; f < 4; ++f) {
    v16h af = load_frag16(A, mbase + f * 16, KP3, 0, lane);
    acc[f] = __builtin_amdgcn_wmma_f32_16x16x32_f16(
        false, af, false, bf, (short)0, acc[f], false, false);
  }

  int col  = nbase + (lane & 15);
  int rsel = (lane >> 4) << 3;
  if (col < 392) {
    float bv = bias[col];
#pragma unroll
    for (int f = 0; f < 4; ++f) {
#pragma unroll
      for (int i = 0; i < 8; ++i) {
        int r = mbase + f * 16 + i + rsel;
        float v = acc[f][i] + bv;
        out[(size_t)r * 392 + col] = v > 0.f ? v : 0.f;
      }
    }
  }
}

// ---------------------------------------------------------------------------
// K6: per-sample decoder: conv(2->11)+relu on 14x14, nearest x2 upsample fused
// into conv(11->3)+relu on 28x28, conv(3->1)+sigmoid -> recon.
// ---------------------------------------------------------------------------
__global__ void vae_decode_kernel(const float* __restrict__ h3in, // [B][392]
                                  const float* __restrict__ w0g,  // [11][2][3][3]
                                  const float* __restrict__ w1g,  // [3][11][3][3]
                                  const float* __restrict__ w2g,  // [1][3][3][3]
                                  float* __restrict__ recon) {    // [B][784]
  __shared__ float h3s[392];
  __shared__ float d0s[11 * 196];
  __shared__ float d1s[3 * 784];
  __shared__ float w0[198];
  __shared__ float w1[297];
  __shared__ float w2[27];

  const int b   = blockIdx.x;
  const int tid = threadIdx.x;

  if (tid < 198) w0[tid] = w0g[tid];
  for (int i = tid; i < 297; i += blockDim.x) w1[i] = w1g[i];
  if (tid < 27) w2[tid] = w2g[tid];
  for (int i = tid; i < 392; i += blockDim.x) h3s[i] = h3in[(size_t)b * 392 + i];
  __syncthreads();

  // de_conv0: 2 -> 11 on 14x14, relu
  for (int oi = tid; oi < 11 * 196; oi += blockDim.x) {
    int o = oi / 196, s = oi % 196, hh = s / 14, ww = s % 14;
    float acc = 0.f;
#pragma unroll
    for (int ic = 0; ic < 2; ++ic) {
      const float* wp = &w0[(o * 2 + ic) * 9];
#pragma unroll
      for (int kh = 0; kh < 3; ++kh) {
        int ih = hh + kh - 1;
        if (ih < 0 || ih >= 14) continue;
#pragma unroll
        for (int kw = 0; kw < 3; ++kw) {
          int iw = ww + kw - 1;
          if (iw < 0 || iw >= 14) continue;
          acc += h3s[ic * 196 + ih * 14 + iw] * wp[kh * 3 + kw];
        }
      }
    }
    d0s[oi] = acc > 0.f ? acc : 0.f;
  }
  __syncthreads();

  // de_conv1 on nearest-x2-upsampled 28x28 input, 11 -> 3, relu
  for (int oi = tid; oi < 3 * 784; oi += blockDim.x) {
    int o = oi / 784, s = oi % 784, hh = s / 28, ww = s % 28;
    float acc = 0.f;
#pragma unroll
    for (int ic = 0; ic < 11; ++ic) {
      const float* wp = &w1[(o * 11 + ic) * 9];
#pragma unroll
      for (int kh = 0; kh < 3; ++kh) {
        int ih = hh + kh - 1;
        if (ih < 0 || ih >= 28) continue;
#pragma unroll
        for (int kw = 0; kw < 3; ++kw) {
          int iw = ww + kw - 1;
          if (iw < 0 || iw >= 28) continue;
          acc += d0s[ic * 196 + (ih >> 1) * 14 + (iw >> 1)] * wp[kh * 3 + kw];
        }
      }
    }
    d1s[oi] = acc > 0.f ? acc : 0.f;
  }
  __syncthreads();

  // de_conv2: 3 -> 1, sigmoid
  float* rb = recon + (size_t)b * 784;
  for (int s = tid; s < 784; s += blockDim.x) {
    int hh = s / 28, ww = s % 28;
    float acc = 0.f;
#pragma unroll
    for (int ic = 0; ic < 3; ++ic) {
      const float* wp = &w2[ic * 9];
#pragma unroll
      for (int kh = 0; kh < 3; ++kh) {
        int ih = hh + kh - 1;
        if (ih < 0 || ih >= 28) continue;
#pragma unroll
        for (int kw = 0; kw < 3; ++kw) {
          int iw = ww + kw - 1;
          if (iw < 0 || iw >= 28) continue;
          acc += d1s[ic * 784 + ih * 28 + iw] * wp[kh * 3 + kw];
        }
      }
    }
    rb[s] = 1.f / (1.f + __expf(-acc));
  }
}

// ---------------------------------------------------------------------------
extern "C" void kernel_launch(void* const* d_in, const int* in_sizes, int n_in,
                              void* d_out, int out_size, void* d_ws, size_t ws_size,
                              hipStream_t stream) {
  const float*     x        = (const float*)d_in[0];
  const long long* cls      = (const long long*)d_in[1];
  const float*     en0_w    = (const float*)d_in[2];
  const float*     en1_w    = (const float*)d_in[3];
  const float*     fc1_w    = (const float*)d_in[4];
  const float*     fc1_b    = (const float*)d_in[5];
  const float*     fc21_w   = (const float*)d_in[6];
  const float*     fc21_b   = (const float*)d_in[7];
  const float*     fc22_w   = (const float*)d_in[8];
  const float*     fc22_b   = (const float*)d_in[9];
  const float*     fc3_w    = (const float*)d_in[10];
  const float*     fc3_b    = (const float*)d_in[11];
  const float*     de0_w    = (const float*)d_in[12];
  const float*     de1_w    = (const float*)d_in[13];
  const float*     de2_w    = (const float*)d_in[14];

  char* ws = (char*)d_ws;
  _Float16* h_act  = (_Float16*)(ws + OFF_H);
  _Float16* w1h    = (_Float16*)(ws + OFF_W1);
  _Float16* h1_act = (_Float16*)(ws + OFF_H1);
  _Float16* w2h    = (_Float16*)(ws + OFF_W2);
  _Float16* z_act  = (_Float16*)(ws + OFF_Z);
  _Float16* w3h    = (_Float16*)(ws + OFF_W3);
  float*    h3buf  = (float*)   (ws + OFF_H3);

  float* out    = (float*)d_out;
  float* recon  = out;                               // [B][784]
  float* mu_out = out + (size_t)BATCH * 784;         // [B][20]
  float* lv_out = mu_out + (size_t)BATCH * 20;       // [B][20]

  const int T = 256;

  // K0: weight conversion / padding
  {
    int total = 400 * KP1 + 48 * KP2 + 400 * KP3 + BATCH * 16;
    vae_prep_kernel<<<(total + T - 1) / T, T, 0, stream>>>(
        fc1_w, fc21_w, fc22_w, fc3_w, w1h, w2h, w3h, h1_act);
  }
  // K1: encoder convs (one block per sample)
  vae_encode_kernel<<<BATCH, T, 0, stream>>>(x, cls, en0_w, en1_w, h_act);
  // K2: fc1 (WMMA, async-LDS staged A)  128 blocks x 25 waves
  vae_fc1_kernel<<<BATCH / 64, 800, 0, stream>>>(h_act, w1h, fc1_b, h1_act);
  // K3: fc21/fc22 (WMMA)  384 waves
  vae_fc2_kernel<<<(128 * 3) / 8, T, 0, stream>>>(h1_act, w2h, fc21_b, fc22_b,
                                                  mu_out, lv_out, z_act);
  // K4: z one-hot fill
  vae_zfill_kernel<<<(BATCH * 12 + T - 1) / T, T, 0, stream>>>(cls, z_act);
  // K5: fc3 (WMMA)  3200 waves
  vae_fc3_kernel<<<(128 * 25) / 8, T, 0, stream>>>(z_act, w3h, fc3_b, h3buf);
  // K6: decoder convs (one block per sample)
  vae_decode_kernel<<<BATCH, T, 0, stream>>>(h3buf, de0_w, de1_w, de2_w, recon);

  (void)in_sizes; (void)n_in; (void)out_size; (void)ws_size;
}
